// CausalSelfAttention_10995116278102
// MI455X (gfx1250) — matrix operand order repaired and checked by reading, not compiled, not run on hardware
//
#include <hip/hip_runtime.h>

// ---------------------------------------------------------------------------
// CDNA5 (gfx1250, wave32) causal self-attention:
//   qkv = x @ w_qkv^T          (bf16 WMMA, f32 accum, double-buffered LDS)
//   y   = flash_attn(q, k, v)  (bf16 WMMA, f32 accum, online base-2 softmax)
//   out = y @ w_out^T          (bf16 WMMA, f32 accum, f32 store)
// B=4, T=2048, D=1024, H=16, Dh=64  (all tile sizes divide evenly; no guards)
// ---------------------------------------------------------------------------

typedef __attribute__((ext_vector_type(16))) __bf16       v16bf;
typedef __attribute__((ext_vector_type(8)))  float        v8f;
typedef __attribute__((ext_vector_type(4)))  unsigned int u32x4;

union Frag { v16bf v; u32x4 q[2]; };

__device__ __forceinline__ unsigned short f32_bf16(float f) {
  union { float f; unsigned int u; } c; c.f = f;
  unsigned int u = c.u;
  return (unsigned short)((u + 0x7FFFu + ((u >> 16) & 1u)) >> 16); // RNE
}

// ---------------------------------------------------------------------------
// Kernel 1: f32 -> bf16 conversion (grid-stride)
// ---------------------------------------------------------------------------
__global__ __launch_bounds__(256) void cvt_bf16(const float* __restrict__ src,
                                                unsigned short* __restrict__ dst,
                                                int n) {
  int i = blockIdx.x * blockDim.x + threadIdx.x;
  int stride = gridDim.x * blockDim.x;
  for (; i < n; i += stride) dst[i] = f32_bf16(src[i]);
}

// ---------------------------------------------------------------------------
// Kernel 2/4: C[M,N] = A[M,K] * B[N,K]^T   (bf16 in, f32 accum)
// Block tile 128x128, K-step 32, 256 threads = 8 waves, each wave 64x32.
// Double-buffered LDS: next tile's global loads overlap current tile's WMMAs.
// ---------------------------------------------------------------------------
template <bool STORE_F32>
__global__ __launch_bounds__(256) void gemm_tn(const unsigned short* __restrict__ A,
                                               const unsigned short* __restrict__ B,
                                               void* __restrict__ Cv,
                                               int M, int N, int K) {
  __shared__ __align__(16) unsigned short sA[2][128][40]; // 32 + 8 pad
  __shared__ __align__(16) unsigned short sB[2][128][40]; // stored [n][k]

  const int tid  = threadIdx.x;
  const int lane = tid & 31;
  const int wave = tid >> 5;
  const int wm   = wave & 1;   // 64-row slab
  const int wn   = wave >> 1;  // 32-col slab
  const int bM   = blockIdx.x * 128;
  const int bN   = blockIdx.y * 128;

  v8f acc[4][2] = {};

  const int ldr = tid >> 2;        // 0..63
  const int ldc = (tid & 3) * 8;   // 0,8,16,24

  const unsigned short* pA0 = A + (size_t)(bM + ldr) * K + ldc;
  const unsigned short* pA1 = A + (size_t)(bM + ldr + 64) * K + ldc;
  const unsigned short* pB0 = B + (size_t)(bN + ldr) * K + ldc;
  const unsigned short* pB1 = B + (size_t)(bN + ldr + 64) * K + ldc;

  // Prologue: stage K-tile 0 into buffer 0
  {
    u32x4 ra0 = *(const u32x4*)(pA0);
    u32x4 ra1 = *(const u32x4*)(pA1);
    u32x4 rb0 = *(const u32x4*)(pB0);
    u32x4 rb1 = *(const u32x4*)(pB1);
    *(u32x4*)&sA[0][ldr][ldc]      = ra0;
    *(u32x4*)&sA[0][ldr + 64][ldc] = ra1;
    *(u32x4*)&sB[0][ldr][ldc]      = rb0;
    *(u32x4*)&sB[0][ldr + 64][ldc] = rb1;
  }
  __syncthreads();

  const int nsteps = K / 32;
  for (int ks = 0; ks < nsteps; ++ks) {
    const int p = ks & 1;
    const bool more = (ks + 1) < nsteps;

    // Issue next tile's global loads early; they retire under this tile's WMMAs
    u32x4 ra0, ra1, rb0, rb1;
    if (more) {
      const int k1 = (ks + 1) * 32;
      ra0 = *(const u32x4*)(pA0 + k1);
      ra1 = *(const u32x4*)(pA1 + k1);
      rb0 = *(const u32x4*)(pB0 + k1);
      rb1 = *(const u32x4*)(pB1 + k1);
    }

    // A 16x32 bf16 frag: lane<16 row=lane, K{0..7,16..23}; lane>=16 K{8..15,24..31}
    Frag a[4], b[2];
    const int kA = (lane >> 4) * 8;
#pragma unroll
    for (int am = 0; am < 4; ++am) {
      int r = wm * 64 + am * 16 + (lane & 15);
      a[am].q[0] = *(const u32x4*)&sA[p][r][kA];
      a[am].q[1] = *(const u32x4*)&sA[p][r][kA + 16];
    }
    // B 32x16 bf16 frag: lane holds column n=lane&15, K{0..7,16..23} / K{8..15,24..31}
    const int kB = (lane >> 4) * 8;
#pragma unroll
    for (int bn = 0; bn < 2; ++bn) {
      int r = wn * 32 + bn * 16 + (lane & 15);
      b[bn].q[0] = *(const u32x4*)&sB[p][r][kB];
      b[bn].q[1] = *(const u32x4*)&sB[p][r][kB + 16];
    }
#pragma unroll
    for (int am = 0; am < 4; ++am)
#pragma unroll
      for (int bn = 0; bn < 2; ++bn)
        acc[am][bn] = __builtin_amdgcn_wmma_f32_16x16x32_bf16(
            false, a[am].v, false, b[bn].v, (short)0, acc[am][bn], false, false);

    // Stage next tile into the other buffer (safe: last readers of it passed
    // the barrier that ended iteration ks-1), then one barrier per iteration.
    if (more) {
      *(u32x4*)&sA[p ^ 1][ldr][ldc]      = ra0;
      *(u32x4*)&sA[p ^ 1][ldr + 64][ldc] = ra1;
      *(u32x4*)&sB[p ^ 1][ldr][ldc]      = rb0;
      *(u32x4*)&sB[p ^ 1][ldr + 64][ldc] = rb1;
    }
    __syncthreads();
  }

  // C layout: vgpr v -> row v + 8*(lane>=16), col = lane&15
  const int mo = (lane >> 4) * 8;
  const int nl = lane & 15;
#pragma unroll
  for (int am = 0; am < 4; ++am)
#pragma unroll
    for (int bn = 0; bn < 2; ++bn)
#pragma unroll
      for (int v = 0; v < 8; ++v) {
        size_t m = (size_t)(bM + wm * 64 + am * 16 + mo + v);
        size_t n = (size_t)(bN + wn * 32 + bn * 16 + nl);
        float f = acc[am][bn][v];
        if (STORE_F32) ((float*)Cv)[m * N + n] = f;
        else           ((unsigned short*)Cv)[m * N + n] = f32_bf16(f);
      }
}

// ---------------------------------------------------------------------------
// Kernel 3: flash attention. Block = 64 queries of one (b,h); 4 waves x 16 rows.
// qkv layout per token row (3072 bf16): [q(0..1023) | k(1024..2047) | v(2048..3071)]
// Softmax runs in the base-2 domain: SCL = Dh^-0.5 * log2(e) folded into S.
// ---------------------------------------------------------------------------
__global__ __launch_bounds__(128) void flash_attn(const unsigned short* __restrict__ qkv,
                                                  unsigned short* __restrict__ y) {
  __shared__ __align__(16) unsigned short sVT[64][72];    // V^T: [d][key], per 64-key block
  __shared__ __align__(16) unsigned short sP[4][16][72];  // per-wave P relayout buffer

  const int T = 2048, E = 3072, HD = 64;
  const float SCL = 0.125f * 1.44269504088896340736f;  // Dh^-0.5 * log2(e)
  const int lane = threadIdx.x & 31;
  const int wave = threadIdx.x >> 5;
  const int qb = blockIdx.x;            // 0..31 query block
  const int bh = blockIdx.y;            // 0..63
  const int b = bh >> 4, h = bh & 15;
  const size_t rowbase = (size_t)b * T;
  const int qr0 = qb * 64 + wave * 16;  // this wave's first query row

  const int mo = (lane >> 4) * 8;
  const int nl = lane & 15;

  // Q fragments: 16x64 -> two 16x32 bf16 A-frags, straight from global
  Frag qf[2];
  {
    int t = qr0 + nl;
    const unsigned short* qp = qkv + (rowbase + t) * E + h * HD;
#pragma unroll
    for (int c = 0; c < 2; ++c) {
      int ko = c * 32 + ((lane >> 4) << 3);
      qf[c].q[0] = *(const u32x4*)(qp + ko);
      qf[c].q[1] = *(const u32x4*)(qp + ko + 16);
    }
  }

  v8f yacc[4] = {};
  float mrow[8], lrow[8];
#pragma unroll
  for (int i = 0; i < 8; ++i) { mrow[i] = -1e30f; lrow[i] = 0.f; }

  for (int kb = 0; kb <= qb; ++kb) {   // causal: all waves same trip count
    const int k0 = kb * 64;
    __syncthreads();                   // protect sVT overwrite vs prev readers
    // Stage V^T (64 keys x 64 dims) into LDS, cooperatively
    {
      int j  = threadIdx.x >> 1;            // key 0..63
      int d0 = (threadIdx.x & 1) * 32;      // half of head dim
      const unsigned short* vp = qkv + (rowbase + k0 + j) * E + 2048 + h * HD + d0;
      alignas(16) unsigned short tmp[32];
      *(u32x4*)&tmp[0]  = *(const u32x4*)(vp);
      *(u32x4*)&tmp[8]  = *(const u32x4*)(vp + 8);
      *(u32x4*)&tmp[16] = *(const u32x4*)(vp + 16);
      *(u32x4*)&tmp[24] = *(const u32x4*)(vp + 24);
#pragma unroll
      for (int d = 0; d < 32; ++d) sVT[d0 + d][j] = tmp[d];
    }
    __syncthreads();                   // sVT visible to all waves

    // S = Q K^T over 64 keys: 4 tiles of 16x16, each 2 WMMAs over Dh=64
    v8f s[4] = {};
#pragma unroll
    for (int nt = 0; nt < 4; ++nt) {
#pragma unroll
      for (int c = 0; c < 2; ++c) {
        Frag kf;  // B-frag column = key row of K -> contiguous in memory
        int key = k0 + nt * 16 + nl;
        const unsigned short* kp = qkv + (rowbase + key) * E + 1024 + h * HD
                                   + c * 32 + ((lane >> 4) << 3);
        kf.q[0] = *(const u32x4*)(kp);
        kf.q[1] = *(const u32x4*)(kp + 16);
        s[nt] = __builtin_amdgcn_wmma_f32_16x16x32_bf16(
            false, qf[c].v, false, kf.v, (short)0, s[nt], false, false);
      }
    }

    // Scale (base-2 domain) + causal mask + online softmax.
    // Row m of a C tile = vgpr index v + 8*(lane>=16); col = lane&15.
#pragma unroll
    for (int v = 0; v < 8; ++v) {
      const int qg = qr0 + mo + v;
      float mx = -1e30f;
#pragma unroll
      for (int nt = 0; nt < 4; ++nt) {
        const int kg = k0 + nt * 16 + nl;
        float e = (kg > qg) ? -1e30f : s[nt][v] * SCL;
        s[nt][v] = e;
        mx = fmaxf(mx, e);
      }
#pragma unroll
      for (int off = 1; off < 16; off <<= 1)
        mx = fmaxf(mx, __shfl_xor(mx, off, 32));
      const float mnew  = fmaxf(mrow[v], mx);
      const float alpha = exp2f(mrow[v] - mnew);
      float sum = 0.f;
#pragma unroll
      for (int nt = 0; nt < 4; ++nt) {
        float p = exp2f(s[nt][v] - mnew);
        s[nt][v] = p;
        sum += p;
      }
#pragma unroll
      for (int off = 1; off < 16; off <<= 1)
        sum += __shfl_xor(sum, off, 32);
      lrow[v] = lrow[v] * alpha + sum;
      mrow[v] = mnew;
#pragma unroll
      for (int dt = 0; dt < 4; ++dt) yacc[dt][v] *= alpha;
    }

    // Re-layout P from C layout to A layout through this wave's private LDS
    // buffer. No barrier needed: same-wave LDS RAW is ordered by DScnt waits.
#pragma unroll
    for (int nt = 0; nt < 4; ++nt)
#pragma unroll
      for (int v = 0; v < 8; ++v)
        sP[wave][mo + v][nt * 16 + nl] = f32_bf16(s[nt][v]);

    Frag pf[2];
#pragma unroll
    for (int c = 0; c < 2; ++c) {
      int kk = c * 32 + ((lane >> 4) << 3);
      pf[c].q[0] = *(const u32x4*)&sP[wave][nl][kk];
      pf[c].q[1] = *(const u32x4*)&sP[wave][nl][kk + 16];
    }
    // Y += P (16x64) x V (64x16 per d-tile); V^T in LDS gives contiguous columns
#pragma unroll
    for (int dt = 0; dt < 4; ++dt) {
#pragma unroll
      for (int c = 0; c < 2; ++c) {
        Frag vf;
        int ks = c * 32 + ((lane >> 4) << 3);
        vf.q[0] = *(const u32x4*)&sVT[dt * 16 + nl][ks];
        vf.q[1] = *(const u32x4*)&sVT[dt * 16 + nl][ks + 16];
        yacc[dt] = __builtin_amdgcn_wmma_f32_16x16x32_bf16(
            false, pf[c].v, false, vf.v, (short)0, yacc[dt], false, false);
      }
    }
  }

  // Epilogue: one reciprocal per row, bf16 store to [B*T, H*Dh]
#pragma unroll
  for (int v = 0; v < 8; ++v) {
    const float rinv = __frcp_rn(lrow[v]);
    const int t = qr0 + mo + v;
    unsigned short* yp = y + (rowbase + t) * 1024 + h * HD + nl;
#pragma unroll
    for (int dt = 0; dt < 4; ++dt)
      yp[dt * 16] = f32_bf16(yacc[dt][v] * rinv);
  }
}

// ---------------------------------------------------------------------------
// Host launch
// ---------------------------------------------------------------------------
extern "C" void kernel_launch(void* const* d_in, const int* in_sizes, int n_in,
                              void* d_out, int out_size, void* d_ws, size_t ws_size,
                              hipStream_t stream) {
  (void)in_sizes; (void)n_in; (void)out_size; (void)ws_size;
  const float* x     = (const float*)d_in[0];  // [4,2048,1024]
  const float* w_qkv = (const float*)d_in[1];  // [3072,1024]
  const float* w_out = (const float*)d_in[2];  // [1024,1024]
  float* out = (float*)d_out;                  // [4,2048,1024] f32

  const int BT = 4 * 2048; // 8192 token rows

  // Workspace layout (bf16 buffers), total ~92.3 MB
  unsigned short* xb   = (unsigned short*)d_ws;
  unsigned short* wqb  = xb   + (size_t)BT * 1024;
  unsigned short* wob  = wqb  + (size_t)3072 * 1024;
  unsigned short* qkvb = wob  + (size_t)1024 * 1024;
  unsigned short* yb   = qkvb + (size_t)BT * 3072;

  cvt_bf16<<<512, 256, 0, stream>>>(x,     xb,  BT * 1024);
  cvt_bf16<<<256, 256, 0, stream>>>(w_qkv, wqb, 3072 * 1024);
  cvt_bf16<<<128, 256, 0, stream>>>(w_out, wob, 1024 * 1024);

  // qkv[b,t,e] = sum_d x[b,t,d] * w_qkv[e,d]
  gemm_tn<false><<<dim3(BT / 128, 3072 / 128), 256, 0, stream>>>(
      xb, wqb, qkvb, BT, 3072, 1024);

  // flash attention: grid = (query blocks of 64) x (B*H)
  flash_attn<<<dim3(2048 / 64, 4 * 16), 128, 0, stream>>>(qkvb, yb);

  // out[b,t,d] = sum_e y[b,t,e] * w_out[d,e]  (f32 store)
  gemm_tn<true><<<dim3(BT / 128, 1024 / 128), 256, 0, stream>>>(
      yb, wob, out, BT, 1024, 1024);
}
